// DenseMotionNetwork_70205535420833
// MI455X (gfx1250) — matrix operand order repaired and checked by reading, not compiled, not run on hardware
//
#include <hip/hip_runtime.h>

#define BSZ 32
#define KP  10
#define NCH 3

typedef _Float16 h16;
typedef __attribute__((ext_vector_type(16))) _Float16 v16h;
typedef __attribute__((ext_vector_type(8)))  _Float16 v8h;
typedef __attribute__((ext_vector_type(8)))  float    v8f;

// ---------------------------------------------------------------------------
// T = ksj * inv(kdj) per (batch, keypoint)
// ---------------------------------------------------------------------------
__global__ void tmat_kernel(const float* __restrict__ kdj, const float* __restrict__ ksj,
                            float* __restrict__ tmat) {
  int i = blockIdx.x * 64 + threadIdx.x;
  if (i >= BSZ * KP) return;
  const float* J = kdj + i * 4;
  const float* S = ksj + i * 4;
  float r = 1.0f / (J[0] * J[3] - J[1] * J[2]);
  float i0 =  J[3] * r, i1 = -J[1] * r, i2 = -J[2] * r, i3 = J[0] * r;
  tmat[i * 4 + 0] = S[0] * i0 + S[1] * i2;
  tmat[i * 4 + 1] = S[0] * i1 + S[1] * i3;
  tmat[i * 4 + 2] = S[2] * i0 + S[3] * i2;
  tmat[i * 4 + 3] = S[2] * i1 + S[3] * i3;
}

__device__ __forceinline__ void sparse_coord(int b, int kp, float gx, float gy,
                                             const float* __restrict__ kdv,
                                             const float* __restrict__ ksv,
                                             const float* __restrict__ tmat,
                                             float& qx, float& qy) {
  if (kp == 0) { qx = gx; qy = gy; return; }
  int k = kp - 1;
  const float* kd = kdv + (b * KP + k) * 2;
  const float* ks = ksv + (b * KP + k) * 2;
  const float* T  = tmat + (b * KP + k) * 4;
  float dx = gx - kd[0], dy = gy - kd[1];
  qx = T[0] * dx + T[1] * dy + ks[0];
  qy = T[2] * dx + T[3] * dy + ks[1];
}

// ---------------------------------------------------------------------------
// Deformed source images: per (b, kp) bilinear grid_sample (zeros padding,
// align_corners=False) of the 3-channel source.  Writes d_out region 0.
// ---------------------------------------------------------------------------
__global__ void deform_kernel(const float* __restrict__ img, const float* __restrict__ kdv,
                              const float* __restrict__ ksv, const float* __restrict__ tmat,
                              float* __restrict__ outDef) {
  int i = blockIdx.x * 256 + threadIdx.x;
  if (i >= BSZ * (KP + 1) * 4096) return;
  int hw = i & 4095;
  int x = hw & 63, y = hw >> 6;
  int bk = i >> 12;
  int kp = bk % (KP + 1);
  int b  = bk / (KP + 1);
  float gx = -1.f + x * (2.f / 63.f);
  float gy = -1.f + y * (2.f / 63.f);
  float qx, qy;
  sparse_coord(b, kp, gx, gy, kdv, ksv, tmat, qx, qy);
  float fx = (qx + 1.f) * 32.f - 0.5f;
  float fy = (qy + 1.f) * 32.f - 0.5f;
  float flx = floorf(fx), fly = floorf(fy);
  int x0 = (int)flx, y0 = (int)fly;
  int x1 = x0 + 1,  y1 = y0 + 1;
  float wx = fx - flx, wy = fy - fly;
  bool vx0 = (unsigned)x0 < 64u, vx1 = (unsigned)x1 < 64u;
  bool vy0 = (unsigned)y0 < 64u, vy1 = (unsigned)y1 < 64u;
  float w00 = (1.f - wx) * (1.f - wy), w10 = wx * (1.f - wy);
  float w01 = (1.f - wx) * wy,         w11 = wx * wy;
  const float* ib = img + (size_t)b * NCH * 4096;
  float* ob = outDef + (size_t)bk * NCH * 4096 + hw;
  for (int c = 0; c < NCH; ++c) {
    const float* ic = ib + c * 4096;
    float v = 0.f;
    if (vx0 && vy0) v += w00 * ic[y0 * 64 + x0];
    if (vx1 && vy0) v += w10 * ic[y0 * 64 + x1];
    if (vx0 && vy1) v += w01 * ic[y1 * 64 + x0];
    if (vx1 && vy1) v += w11 * ic[y1 * 64 + x1];
    ob[(size_t)c * 4096] = v;
  }
}

// ---------------------------------------------------------------------------
// Build 44-channel hourglass input (heat | deformed), f32 -> f16.
// Channel ch = kp*4 + c; c==0 -> heatmap, c=1..3 -> deformed RGB.
// ---------------------------------------------------------------------------
__global__ void build_inp_kernel(const float* __restrict__ deformed,
                                 const float* __restrict__ kdv, const float* __restrict__ ksv,
                                 h16* __restrict__ inp) {
  int i = blockIdx.x * 256 + threadIdx.x;
  if (i >= BSZ * 44 * 4096) return;
  int hw = i & 4095;
  int x = hw & 63, y = hw >> 6;
  int bc = i >> 12;
  int ch = bc % 44;
  int b  = bc / 44;
  int kp = ch >> 2, c = ch & 3;
  float v;
  if (c == 0) {
    if (kp == 0) v = 0.f;
    else {
      int k = kp - 1;
      float gx = -1.f + x * (2.f / 63.f);
      float gy = -1.f + y * (2.f / 63.f);
      const float* kd = kdv + (b * KP + k) * 2;
      const float* ks = ksv + (b * KP + k) * 2;
      float dxd = gx - kd[0], dyd = gy - kd[1];
      float dxs = gx - ks[0], dys = gy - ks[1];
      v = expf(-0.5f * (dxd * dxd + dyd * dyd) * 100.f)
        - expf(-0.5f * (dxs * dxs + dys * dys) * 100.f);
    }
  } else {
    v = deformed[(((size_t)b * (KP + 1) + kp) * NCH + (c - 1)) * 4096 + hw];
  }
  inp[i] = (h16)v;
}

// ---------------------------------------------------------------------------
// Weight packing: OIHW f32 -> [Mpad][Kpad] f16 rows (zero padded).
// ---------------------------------------------------------------------------
__global__ void pack_w_kernel(const float* __restrict__ w, h16* __restrict__ wp,
                              int Cout, int Kraw, int Kpad, int Mpad) {
  int i = blockIdx.x * 256 + threadIdx.x;
  int total = Mpad * Kpad;
  if (i >= total) return;
  int m = i / Kpad;
  int k = i - m * Kpad;
  float v = (m < Cout && k < Kraw) ? w[(size_t)m * Kraw + k] : 0.f;
  wp[i] = (h16)v;
}

// Combined mask(11) + occlusion(1) 7x7 weights -> 12 GEMM rows.
__global__ void pack_mo_kernel(const float* __restrict__ mw, const float* __restrict__ ow,
                               h16* __restrict__ wp, int Kraw, int Kpad, int Mpad) {
  int i = blockIdx.x * 256 + threadIdx.x;
  int total = Mpad * Kpad;
  if (i >= total) return;
  int m = i / Kpad;
  int k = i - m * Kpad;
  float v = 0.f;
  if (k < Kraw) {
    if (m < 11)       v = mw[(size_t)m * Kraw + k];
    else if (m == 11) v = ow[k];
  }
  wp[i] = (h16)v;
}

__global__ void bias_mo_kernel(const float* __restrict__ mb, const float* __restrict__ ob,
                               float* __restrict__ out) {
  int i = threadIdx.x;
  if (i < 16) out[i] = (i < 11) ? mb[i] : ((i == 11) ? ob[0] : 0.f);
}

// ---------------------------------------------------------------------------
// Implicit-GEMM convolution with WMMA f16->f32 accumulate.
//   GEMM: D[Cout][N] = Wp[Cout][K] * B[K][N],  N = BSZ*H*W,  K = Cin*ks^2
//   Block = 256 thr = 8 waves laid out 4(M) x 2(N); block tile 128x64.
//   Each wave owns a 32x32 sub-tile: 2x2 accumulators -> 4 independent
//   v_wmma_f32_16x16x32_f16 per K-chunk (A/B fragments reused across them).
//   ups=1 fuses nearest x2 upsample into the im2col gather.
//   mode 0: y = relu(g*(acc+bias)+be) -> f16 ;  mode 1: y = acc+bias -> f32.
// ---------------------------------------------------------------------------
__global__ __launch_bounds__(256)
void conv_wmma(const h16* __restrict__ act, const h16* __restrict__ wp,
               const float* __restrict__ bias, const float* __restrict__ gamma,
               const float* __restrict__ beta, void* __restrict__ dst,
               int Cin, int Cout, int CoutStride, int shiftW,
               int ks, int pad, int ups, int Kraw, int Kpad, int mode) {
  __shared__ h16 lA[128 * 40];  // 128 M-rows x 32 K, pitch 40 (bank-conflict free)
  __shared__ h16 lB[64 * 40];   // 64 N-rows (B^T) x 32 K

  const int t    = threadIdx.x;
  const int lane = t & 31;
  const int wv   = t >> 5;
  const int wm   = wv & 3;      // 4 waves along M
  const int wn   = wv >> 2;     // 2 waves along N
  const int mBlk = blockIdx.y * 128;
  const int nBlk = blockIdx.x * 64;

  const int W    = 1 << shiftW;          // output spatial (H==W)
  const int HWs  = 2 * shiftW;
  const int HWm1 = (1 << HWs) - 1;
  const int Ws   = W >> ups;             // input (source) spatial
  const int ks2  = ks * ks;

  // A staging: thread -> (row, two 8-half chunks)
  const int arow  = t >> 1;              // 0..127
  const int akoff = (t & 1) * 16;        // 0 or 16
  // B staging: thread -> (pixel, 8 k-values)
  const int bpix  = t >> 2;              // 0..63
  const int bkoff = (t & 3) * 8;         // 0,8,16,24
  const int ng    = nBlk + bpix;
  const int gb    = ng >> HWs;
  const int ghw   = ng & HWm1;
  const int gy    = ghw >> shiftW;
  const int gx    = ghw & (W - 1);

  // Fragment addressing (wave32), this code's contiguous k order, same for A and B:
  //  A 16x32: lane row = lane%16; VGPR0-7 = K[hi*16 .. +15]
  //  B 32x16: lane col = lane%16; VGPR0-7 = K[hi*16 .. +15]
  const int fr = lane & 15;
  const int hi = lane >> 4;
  const h16* aRow0 = &lA[(wm * 32 + fr) * 40];
  const h16* aRow1 = &lA[(wm * 32 + 16 + fr) * 40];
  const h16* bRow0 = &lB[(wn * 32 + fr) * 40];
  const h16* bRow1 = &lB[(wn * 32 + 16 + fr) * 40];
  const int ak0 = hi * 16, ak1 = hi * 16 + 8;
  const int bk0 = hi * 16;

  v8f acc00 = {}, acc01 = {}, acc10 = {}, acc11 = {};
  const int nK = Kpad >> 5;
  for (int kc = 0; kc < nK; ++kc) {
    __syncthreads();
    // ---- stage A (packed f16 weights): 128x32 halfs, 2 x b128 per thread ----
    {
      const h16* s = wp + (size_t)(mBlk + arow) * Kpad + kc * 32 + akoff;
      *(v8h*)&lA[arow * 40 + akoff]     = *(const v8h*)s;
      *(v8h*)&lA[arow * 40 + akoff + 8] = *(const v8h*)(s + 8);
      if (kc + 1 < nK) __builtin_prefetch(s + 32, 0, 1);   // global_prefetch_b8
    }
    // ---- stage B (im2col gather, fused upsample + zero pad): 64x32 halfs ----
    {
      int kg  = kc * 32 + bkoff;
      int ci  = kg / ks2;
      int rem = kg - ci * ks2;
      int ky  = rem / ks;
      int kx  = rem - ky * ks;
#pragma unroll
      for (int j = 0; j < 8; ++j) {
        h16 v = (h16)0.f;
        if (kg + j < Kraw) {
          int iy = gy + ky - pad;
          int ix = gx + kx - pad;
          if ((unsigned)iy < (unsigned)W && (unsigned)ix < (unsigned)W) {
            int siy = iy >> ups, six = ix >> ups;
            v = act[((size_t)(gb * Cin + ci) * Ws + siy) * Ws + six];
          }
        }
        lB[bpix * 40 + bkoff + j] = v;
        if (++kx == ks) { kx = 0; if (++ky == ks) { ky = 0; ++ci; } }
      }
    }
    __syncthreads();
    // ---- fragments (b128 LDS loads) + 4 independent WMMAs ----
    v8h a00 = *(const v8h*)(aRow0 + ak0);
    v8h a01 = *(const v8h*)(aRow0 + ak1);
    v8h a10 = *(const v8h*)(aRow1 + ak0);
    v8h a11 = *(const v8h*)(aRow1 + ak1);
    v8h b00 = *(const v8h*)(bRow0 + bk0);
    v8h b01 = *(const v8h*)(bRow0 + bk0 + 8);
    v8h b10 = *(const v8h*)(bRow1 + bk0);
    v8h b11 = *(const v8h*)(bRow1 + bk0 + 8);
    v16h aF0 = __builtin_shufflevector(a00, a01, 0,1,2,3,4,5,6,7,8,9,10,11,12,13,14,15);
    v16h aF1 = __builtin_shufflevector(a10, a11, 0,1,2,3,4,5,6,7,8,9,10,11,12,13,14,15);
    v16h bF0 = __builtin_shufflevector(b00, b01, 0,1,2,3,4,5,6,7,8,9,10,11,12,13,14,15);
    v16h bF1 = __builtin_shufflevector(b10, b11, 0,1,2,3,4,5,6,7,8,9,10,11,12,13,14,15);
    acc00 = __builtin_amdgcn_wmma_f32_16x16x32_f16(false, aF0, false, bF0, (short)0, acc00, false, false);
    acc01 = __builtin_amdgcn_wmma_f32_16x16x32_f16(false, aF0, false, bF1, (short)0, acc01, false, false);
    acc10 = __builtin_amdgcn_wmma_f32_16x16x32_f16(false, aF1, false, bF0, (short)0, acc10, false, false);
    acc11 = __builtin_amdgcn_wmma_f32_16x16x32_f16(false, aF1, false, bF1, (short)0, acc11, false, false);
  }

  // ---- epilogue: C/D layout -> (co = base + hi*8 + r, n = base + lane%16) ----
#pragma unroll
  for (int mi = 0; mi < 2; ++mi) {
#pragma unroll
    for (int ni = 0; ni < 2; ++ni) {
      v8f acc = (mi == 0) ? (ni == 0 ? acc00 : acc01) : (ni == 0 ? acc10 : acc11);
      const int nOut = nBlk + wn * 32 + ni * 16 + fr;
      const int ob   = nOut >> HWs;
      const int ohw  = nOut & HWm1;
#pragma unroll
      for (int r = 0; r < 8; ++r) {
        int co = mBlk + wm * 32 + mi * 16 + hi * 8 + r;
        if (co < Cout) {
          float v = acc[r] + bias[co];
          if (mode == 0) {
            v = gamma[co] * v + beta[co];
            v = v > 0.f ? v : 0.f;
            ((h16*)dst)[((size_t)ob * CoutStride + co) * (size_t)(HWm1 + 1) + ohw] = (h16)v;
          } else {
            ((float*)dst)[((size_t)ob * CoutStride + co) * (size_t)(HWm1 + 1) + ohw] = v;
          }
        }
      }
    }
  }
}

// ---------------------------------------------------------------------------
// 2x2 average pool (stride 2), f16.
// ---------------------------------------------------------------------------
__global__ void avgpool_kernel(const h16* __restrict__ src, h16* __restrict__ dst,
                               int C, int shiftWo) {
  int Wo = 1 << shiftWo;
  int total = (BSZ * C) << (2 * shiftWo);
  int i = blockIdx.x * 256 + threadIdx.x;
  if (i >= total) return;
  int xo = i & (Wo - 1);
  int yo = (i >> shiftWo) & (Wo - 1);
  int bc = i >> (2 * shiftWo);
  int Wi = Wo * 2;
  const h16* s = src + (size_t)bc * Wi * Wi;
  float v = (float)s[(2 * yo) * Wi + 2 * xo]     + (float)s[(2 * yo) * Wi + 2 * xo + 1]
          + (float)s[(2 * yo + 1) * Wi + 2 * xo] + (float)s[(2 * yo + 1) * Wi + 2 * xo + 1];
  dst[i] = (h16)(0.25f * v);
}

// Copy skip connection into the tail channels of a concat buffer.
__global__ void concat_kernel(const h16* __restrict__ src, h16* __restrict__ dst,
                              int Csrc, int Cdst, int coff, int HW) {
  int total = BSZ * Csrc * HW;
  int i = blockIdx.x * 256 + threadIdx.x;
  if (i >= total) return;
  int hw = i % HW;
  int bc = i / HW;
  int c  = bc % Csrc;
  int b  = bc / Csrc;
  dst[((size_t)b * Cdst + coff + c) * HW + hw] = src[i];
}

// ---------------------------------------------------------------------------
// Softmax(11) -> mask, sigmoid -> occlusion, mask-weighted sparse-motion sum
// -> deformation (sparse recomputed analytically; no materialized tensor).
// ---------------------------------------------------------------------------
__global__ void final_kernel(const float* __restrict__ logits, const float* __restrict__ kdv,
                             const float* __restrict__ ksv, const float* __restrict__ tmat,
                             float* __restrict__ outMask, float* __restrict__ outDefo,
                             float* __restrict__ outOcc) {
  int i = blockIdx.x * 256 + threadIdx.x;
  if (i >= BSZ * 4096) return;
  int hw = i & 4095;
  int x = hw & 63, y = hw >> 6;
  int b = i >> 12;
  const float* lb = logits + (size_t)b * 12 * 4096 + hw;
  float l[12];
#pragma unroll
  for (int c = 0; c < 12; ++c) l[c] = lb[(size_t)c * 4096];
  float mx = l[0];
#pragma unroll
  for (int c = 1; c <= 10; ++c) mx = fmaxf(mx, l[c]);
  float e[11], s = 0.f;
#pragma unroll
  for (int c = 0; c <= 10; ++c) { e[c] = expf(l[c] - mx); s += e[c]; }
  float inv = 1.f / s;
  float gx = -1.f + x * (2.f / 63.f);
  float gy = -1.f + y * (2.f / 63.f);
  float dx = 0.f, dy = 0.f;
#pragma unroll
  for (int kp = 0; kp <= 10; ++kp) {
    float m = e[kp] * inv;
    outMask[((size_t)b * (KP + 1) + kp) * 4096 + hw] = m;
    float qx, qy;
    sparse_coord(b, kp, gx, gy, kdv, ksv, tmat, qx, qy);
    dx += m * qx; dy += m * qy;
  }
  outDefo[(size_t)i * 2 + 0] = dx;
  outDefo[(size_t)i * 2 + 1] = dy;
  outOcc[i] = 1.f / (1.f + expf(-l[11]));
}

// ---------------------------------------------------------------------------
extern "C" void kernel_launch(void* const* d_in, const int* in_sizes, int n_in,
                              void* d_out, int out_size, void* d_ws, size_t ws_size,
                              hipStream_t stream) {
  (void)in_sizes; (void)n_in; (void)out_size; (void)ws_size;
  const float* img = (const float*)d_in[0];
  const float* kdv = (const float*)d_in[1];
  const float* kdj = (const float*)d_in[2];
  const float* ksv = (const float*)d_in[3];
  const float* ksj = (const float*)d_in[4];
  const float *dw[5], *db[5], *dg[5], *dbe[5], *uw[5], *ub[5], *ug[5], *ube[5];
  for (int i = 0; i < 5; ++i) {              // params dict insertion order
    int base = 5 + 8 * i;
    dw[i]  = (const float*)d_in[base + 0];
    db[i]  = (const float*)d_in[base + 1];
    dg[i]  = (const float*)d_in[base + 2];
    dbe[i] = (const float*)d_in[base + 3];
    uw[i]  = (const float*)d_in[base + 4];
    ub[i]  = (const float*)d_in[base + 5];
    ug[i]  = (const float*)d_in[base + 6];
    ube[i] = (const float*)d_in[base + 7];
  }
  const float* mask_w = (const float*)d_in[45];
  const float* mask_b = (const float*)d_in[46];
  const float* occ_w  = (const float*)d_in[47];
  const float* occ_b  = (const float*)d_in[48];

  float* out     = (float*)d_out;
  float* outDef  = out;                 // [32,11,3,64,64]
  float* outMask = out + 4325376;       // [32,11,64,64]
  float* outDefo = out + 5767168;       // [32,64,64,2]
  float* outOcc  = out + 6029312;       // [32,1,64,64]

  // ---- workspace layout (~146 MB) ----
  char* ws = (char*)d_ws;
  size_t off = 0;
  auto alloc = [&](size_t bytes) -> char* {
    char* p = ws + off;
    off += (bytes + 255) & ~(size_t)255;
    return p;
  };
  float* tmat   = (float*)alloc((size_t)BSZ * KP * 4 * sizeof(float));
  float* bias12 = (float*)alloc(16 * sizeof(float));
  h16* inp44 = (h16*)alloc((size_t)BSZ * 44   * 4096 * 2);
  h16* h0    = (h16*)alloc((size_t)BSZ * 128  * 1024 * 2);
  h16* h1    = (h16*)alloc((size_t)BSZ * 256  * 256  * 2);
  h16* h2    = (h16*)alloc((size_t)BSZ * 512  * 64   * 2);
  h16* h3    = (h16*)alloc((size_t)BSZ * 1024 * 16   * 2);
  h16* h4    = (h16*)alloc((size_t)BSZ * 1024 * 4    * 2);
  h16* S     = (h16*)alloc((size_t)BSZ * 128  * 4096 * 2);  // pre-pool scratch (max c0)
  h16* cat0  = (h16*)alloc((size_t)BSZ * 2048 * 16   * 2);
  h16* cat1  = (h16*)alloc((size_t)BSZ * 1024 * 64   * 2);
  h16* cat2  = (h16*)alloc((size_t)BSZ * 512  * 256  * 2);
  h16* cat3  = (h16*)alloc((size_t)BSZ * 256  * 1024 * 2);
  h16* pred  = (h16*)alloc((size_t)BSZ * 108  * 4096 * 2);
  h16* wpack = (h16*)alloc((size_t)1024 * 9216 * 2);        // max packed layer (18.9MB)
  float* logits = (float*)alloc((size_t)BSZ * 12 * 4096 * 4);

  // ---- stage 0: jacobian products, deformed source, network input ----
  tmat_kernel<<<(BSZ * KP + 63) / 64, 64, 0, stream>>>(kdj, ksj, tmat);
  {
    int total = BSZ * (KP + 1) * 4096;
    deform_kernel<<<(total + 255) / 256, 256, 0, stream>>>(img, kdv, ksv, tmat, outDef);
  }
  {
    int total = BSZ * 44 * 4096;
    build_inp_kernel<<<(total + 255) / 256, 256, 0, stream>>>(outDef, kdv, ksv, inp44);
  }

  // ---- hourglass down path ----
  const int DIN[5]  = {44, 128, 256, 512, 1024};
  const int DOUT[5] = {128, 256, 512, 1024, 1024};
  const int dsh[5]  = {6, 5, 4, 3, 2};           // conv spatial shift (in==out)
  const h16* dact[5] = {inp44, h0, h1, h2, h3};
  h16* hb[5] = {h0, h1, h2, h3, h4};
  for (int i = 0; i < 5; ++i) {
    int Kraw = DIN[i] * 9;
    int Kpad = ((Kraw + 31) / 32) * 32;
    int Mpad = ((DOUT[i] + 127) / 128) * 128;
    int ptotal = Mpad * Kpad;
    pack_w_kernel<<<(ptotal + 255) / 256, 256, 0, stream>>>(dw[i], wpack, DOUT[i], Kraw, Kpad, Mpad);
    int N = BSZ << (2 * dsh[i]);
    dim3 g(N / 64, Mpad / 128);
    conv_wmma<<<g, 256, 0, stream>>>(dact[i], wpack, db[i], dg[i], dbe[i], S,
                                     DIN[i], DOUT[i], DOUT[i], dsh[i], 3, 1, 0, Kraw, Kpad, 0);
    int so = dsh[i] - 1;
    int ptot = (BSZ * DOUT[i]) << (2 * so);
    avgpool_kernel<<<(ptot + 255) / 256, 256, 0, stream>>>(S, hb[i], DOUT[i], so);
  }

  // ---- hourglass up path (upsample fused into conv gather) ----
  const int UIN[5]  = {1024, 2048, 1024, 512, 256};
  const int UOUT[5] = {1024, 512, 256, 128, 64};
  const int ush[5]  = {2, 3, 4, 5, 6};           // conv OUTPUT spatial shift
  const h16* uact[5]   = {h4, cat0, cat1, cat2, cat3};
  h16* udst[5]         = {cat0, cat1, cat2, cat3, pred};
  const int ustride[5] = {2048, 1024, 512, 256, 108};
  const h16* uskip[5]  = {h3, h2, h1, h0, inp44};
  const int uskipC[5]  = {1024, 512, 256, 128, 44};
  for (int i = 0; i < 5; ++i) {
    int Kraw = UIN[i] * 9;
    int Kpad = ((Kraw + 31) / 32) * 32;
    int Mpad = ((UOUT[i] + 127) / 128) * 128;
    int ptotal = Mpad * Kpad;
    pack_w_kernel<<<(ptotal + 255) / 256, 256, 0, stream>>>(uw[i], wpack, UOUT[i], Kraw, Kpad, Mpad);
    int N = BSZ << (2 * ush[i]);
    dim3 g(N / 64, Mpad / 128);
    conv_wmma<<<g, 256, 0, stream>>>(uact[i], wpack, ub[i], ug[i], ube[i], udst[i],
                                     UIN[i], UOUT[i], ustride[i], ush[i], 3, 1, 1, Kraw, Kpad, 0);
    int HW = 1 << (2 * ush[i]);
    int ctotal = BSZ * uskipC[i] * HW;
    concat_kernel<<<(ctotal + 255) / 256, 256, 0, stream>>>(uskip[i], udst[i],
                                                            uskipC[i], ustride[i], UOUT[i], HW);
  }

  // ---- mask(11) + occlusion(1) as one 7x7 conv, then final fusion ----
  {
    int Kraw = 108 * 49;                         // 5292
    int Kpad = ((Kraw + 31) / 32) * 32;          // 5312
    int Mpad = 128;
    int ptotal = Mpad * Kpad;
    pack_mo_kernel<<<(ptotal + 255) / 256, 256, 0, stream>>>(mask_w, occ_w, wpack, Kraw, Kpad, Mpad);
    bias_mo_kernel<<<1, 16, 0, stream>>>(mask_b, occ_b, bias12);
    int N = BSZ << 12;
    dim3 g(N / 64, 1);
    conv_wmma<<<g, 256, 0, stream>>>(pred, wpack, bias12, nullptr, nullptr, logits,
                                     108, 12, 12, 6, 7, 3, 0, Kraw, Kpad, 1);
    int ftotal = BSZ * 4096;
    final_kernel<<<(ftotal + 255) / 256, 256, 0, stream>>>(logits, kdv, ksv, tmat,
                                                           outMask, outDefo, outOcc);
  }
}
